// TemporalAttentionLoRA_7670811591102
// MI455X (gfx1250) — compile-verified, matrix operand order checked by reading, not run on hardware
//
#include <hip/hip_runtime.h>
#include <hip/hip_bf16.h>

// ---------------------------------------------------------------------------
// TemporalAttention + LoRA, fused for MI455X (gfx1250, wave32, WMMA bf16)
//
// B=16384, T=4, D=1024, H=16, HD=64, R=8, SCALING=2, SCALE=1/8
//
// LoRA folded exactly into dense weights:
//   W1 = W_qkv (+2*Aq@Bq on q cols, +2*Av@Bv on v cols)   [1024 x 3072]
//   W2 = W_proj + 2*Ap@Bp                                  [1024 x 1024]
// GEMMs run on v_wmma_f32_16x16x32_bf16 with fp32 accumulation.
// Register blocking: 6 accumulators/wave (kAttn), 2/wave + LDS-staged B (kProj).
// ---------------------------------------------------------------------------

typedef unsigned int u32;
typedef __attribute__((ext_vector_type(16))) __bf16 v16bf;
typedef __attribute__((ext_vector_type(8)))  float  v8f;

union FragU { uint4 q[2]; v16bf v; };

#define Bn   16384
#define Tn   4
#define Dn   1024
#define Hn   16
#define HDn  64
#define SCALING 2.0f
#define SMSCALE 0.125f   // 1/sqrt(64)

// round-to-nearest-even float -> bf16
__device__ __forceinline__ unsigned short f2bf(float f) {
    u32 u = __builtin_bit_cast(u32, f);
    u32 r = (u + 0x7FFFu + ((u >> 16) & 1u)) >> 16;
    return (unsigned short)r;
}

// k offset inside a 32-wide K fragment for dword j (pair base), lane-half hi
__device__ __forceinline__ int kmap(int j, int hi) {
    return (j < 4 ? j * 2 : 16 + (j - 4) * 2) + hi * 8;
}

// ---------------------------------------------------------------------------
// Kernel 0: fold LoRA into weights, store packed WMMA-fragment layout.
// Fragment = 16 (N) x 32 (K) bf16, 1KB: dword index = lane*8 + j,
//   lane = (n%16) + 16*hi,  dword j holds k-pair {kmap(j,hi), +1}.
// W1 frags: 192*32 = 6144.  W2 frags: 64*32 = 2048. One wave per fragment.
// ---------------------------------------------------------------------------
__global__ __launch_bounds__(256) void kPack(
    const float* __restrict__ Wqkv,  const float* __restrict__ Aq,
    const float* __restrict__ Bq,    const float* __restrict__ Av,
    const float* __restrict__ Bv,    const float* __restrict__ Wproj,
    const float* __restrict__ Ap,    const float* __restrict__ Bp,
    u32* __restrict__ W1p, u32* __restrict__ W2p)
{
    const int wave = threadIdx.x >> 5;
    const int lane = threadIdx.x & 31;
    const int f    = blockIdx.x * 8 + wave;      // 0..8191
    const int hi   = lane >> 4;
    const int nl16 = lane & 15;

    u32 dw[8];
    u32* dst;

    if (f < 6144) {                              // ---- W1 (qkv + lora q/v)
        const int ntile3 = f >> 5;               // 0..191
        const int kt     = f & 31;
        const int n3     = ntile3 * 16 + nl16;   // 0..3071
        const bool isq   = (n3 < 1024);
        const bool isv   = (n3 >= 2048);
        const int  nlora = isq ? n3 : (n3 - 2048);
        const float* Arow = isq ? Aq : Av;
        float brow[8];
        #pragma unroll
        for (int r = 0; r < 8; ++r)
            brow[r] = isq ? Bq[r * 1024 + nlora] : (isv ? Bv[r * 1024 + nlora] : 0.f);

        #pragma unroll
        for (int j = 0; j < 8; ++j) {
            unsigned short h2[2];
            #pragma unroll
            for (int p = 0; p < 2; ++p) {
                const int k = kt * 32 + kmap(j, hi) + p;
                float v = Wqkv[(long)k * 3072 + n3];
                if (isq || isv) {
                    float d = 0.f;
                    #pragma unroll
                    for (int r = 0; r < 8; ++r) d += Arow[k * 8 + r] * brow[r];
                    v += SCALING * d;
                }
                h2[p] = f2bf(v);
            }
            dw[j] = (u32)h2[0] | ((u32)h2[1] << 16);
        }
        dst = W1p + (long)f * 256 + lane * 8;
    } else {                                     // ---- W2 (proj + lora proj)
        const int fw     = f - 6144;
        const int ntile2 = fw >> 5;              // 0..63
        const int kt     = fw & 31;
        const int n      = ntile2 * 16 + nl16;
        float brow[8];
        #pragma unroll
        for (int r = 0; r < 8; ++r) brow[r] = Bp[r * 1024 + n];

        #pragma unroll
        for (int j = 0; j < 8; ++j) {
            unsigned short h2[2];
            #pragma unroll
            for (int p = 0; p < 2; ++p) {
                const int k = kt * 32 + kmap(j, hi) + p;
                float v = Wproj[(long)k * 1024 + n];
                float d = 0.f;
                #pragma unroll
                for (int r = 0; r < 8; ++r) d += Ap[k * 8 + r] * brow[r];
                v += SCALING * d;
                h2[p] = f2bf(v);
            }
            dw[j] = (u32)h2[0] | ((u32)h2[1] << 16);
        }
        dst = W2p + (long)fw * 256 + lane * 8;
    }
    *(uint4*)(dst)     = make_uint4(dw[0], dw[1], dw[2], dw[3]);
    *(uint4*)(dst + 4) = make_uint4(dw[4], dw[5], dw[6], dw[7]);
}

// ---------------------------------------------------------------------------
// Kernel 1: x+pos -> LDS (packed bf16 A-frags), per-head QKV WMMA GEMMs,
// tiny softmax attention, attn-out to HBM as packed bf16 A-frags.
// Block = 256 threads (8 waves), 16 samples = 64 rows. Grid = 1024.
// Wave register block: 2 mtiles x 3 ncols -> 6 accumulators; per k-step
// 2 A-frags + 3 B-frags feed 6 WMMAs.
// LDS: 128KB x-tile + 48KB QKV + 1KB scores (~177KB of 320KB WGP LDS).
// ---------------------------------------------------------------------------
__global__ __launch_bounds__(256) void kAttn(
    const float* __restrict__ x, const float* __restrict__ pos,
    const u32* __restrict__ W1p, u32* __restrict__ AOp)
{
    __shared__ u32   xp[4 * 32 * 256];           // 4 mtiles x 32 ktiles x 1KB
    __shared__ float QKV[3][64][64];             // Q,K,V head tiles (fp32)
    __shared__ float sc[16][4][4];               // scores per sample

    const int  tid  = threadIdx.x;
    const int  wave = tid >> 5;
    const int  lane = tid & 31;
    const long row0 = (long)blockIdx.x * 64;     // global row = b*T + t

    // ---- stage 1: load x + temporal_pos, convert bf16, pack into LDS frags
    for (int it = 0; it < 64; ++it) {
        const int idx  = tid + it * 256;         // 16384 float4 chunks
        const int row  = idx >> 8;               // 0..63
        const int kb   = (idx & 255) * 4;        // col base, multiple of 4
        const float4 xv = *(const float4*)(x + (row0 + row) * Dn + kb);
        const float4 pv = *(const float4*)(pos + (long)(row & 3) * Dn + kb);
        const u32 d0 = (u32)f2bf(xv.x + pv.x) | ((u32)f2bf(xv.y + pv.y) << 16);
        const u32 d1 = (u32)f2bf(xv.z + pv.z) | ((u32)f2bf(xv.w + pv.w) << 16);
        const int mtile = row >> 4, m = row & 15;
        const int ktile = kb >> 5,  kk = kb & 31;
        const int hi = (kk >> 3) & 1;
        const int j0 = ((kk & 7) >> 1) + (kk >= 16 ? 4 : 0);
        const int off = ((mtile * 32 + ktile) << 8) + (m + hi * 16) * 8 + j0;
        *(uint2*)&xp[off] = make_uint2(d0, d1);  // ds_store_b64
    }
    __syncthreads();

    // wave -> register block: mtiles {mbase, mbase+1} x ncols {nbase..nbase+2}
    const int mbase = (wave & 1) * 2;            // 0 or 2
    const int nbase = (wave >> 1) * 3;           // 0,3,6,9 (ncol 0..11)

    // ---- per-head loop
    for (int h = 0; h < Hn; ++h) {
        v8f acc[2][3] = {};
        for (int kt = 0; kt < 32; ++kt) {
            FragU a[2], b[3];
            #pragma unroll
            for (int i = 0; i < 2; ++i) {
                const u32* ap = &xp[(((mbase + i) * 32 + kt) << 8) + lane * 8];
                a[i].q[0] = *(const uint4*)ap;
                a[i].q[1] = *(const uint4*)(ap + 4);
            }
            #pragma unroll
            for (int j = 0; j < 3; ++j) {
                const int ncol = nbase + j;              // 0..11 over {q,k,v}
                const int ntile3 = (ncol >> 2) * 64 + h * 4 + (ncol & 3);
                const uint4* bp =
                    (const uint4*)(W1p + (((long)ntile3 * 32 + kt) << 8)) + lane * 2;
                b[j].q[0] = bp[0];
                b[j].q[1] = bp[1];
                if (kt + 1 < 32)                         // warm WGP$ for next k-step
                    __builtin_prefetch((const void*)(bp + 64), 0, 1);
            }
            #pragma unroll
            for (int i = 0; i < 2; ++i)
                #pragma unroll
                for (int j = 0; j < 3; ++j)
                    acc[i][j] = __builtin_amdgcn_wmma_f32_16x16x32_bf16(
                        false, a[i].v, false, b[j].v, (short)0, acc[i][j], false, false);
        }
        // scatter C tiles (lane L, reg r -> row = r + 8*(L/16), col = L%16)
        #pragma unroll
        for (int i = 0; i < 2; ++i) {
            const int rbase = (mbase + i) * 16 + (lane >> 4) * 8;
            #pragma unroll
            for (int j = 0; j < 3; ++j) {
                const int ncol = nbase + j;
                const int col  = (ncol & 3) * 16 + (lane & 15);
                float* dst = &QKV[ncol >> 2][rbase][col];
                #pragma unroll
                for (int r = 0; r < 8; ++r) dst[r * 64] = acc[i][j][r];
            }
        }
        __syncthreads();

        // scores: 16 samples x (4x4), one entry per thread
        {
            const int s = tid >> 4, idx = tid & 15;
            const int i = idx >> 2, jq = idx & 3;
            const float4* qr = (const float4*)&QKV[0][s * 4 + i][0];
            const float4* kr = (const float4*)&QKV[1][s * 4 + jq][0];
            float dot = 0.f;
            #pragma unroll
            for (int d4 = 0; d4 < 16; ++d4) {
                const float4 aa = qr[d4], bb = kr[d4];
                dot += aa.x * bb.x + aa.y * bb.y + aa.z * bb.z + aa.w * bb.w;
            }
            sc[s][i][jq] = dot * SMSCALE;
        }
        __syncthreads();

        // softmax + attn@V, store bf16 pairs in packed A-frag layout to HBM
        for (int it = 0; it < 8; ++it) {
            const int item  = tid + it * 256;    // 64 rows x 32 col-pairs
            const int row64 = item >> 5;
            const int c0    = (item & 31) * 2;
            const int s = row64 >> 2, i = row64 & 3;
            const float s0 = sc[s][i][0], s1 = sc[s][i][1];
            const float s2 = sc[s][i][2], s3 = sc[s][i][3];
            const float mx = fmaxf(fmaxf(s0, s1), fmaxf(s2, s3));
            const float e0 = __expf(s0 - mx), e1 = __expf(s1 - mx);
            const float e2 = __expf(s2 - mx), e3 = __expf(s3 - mx);
            const float inv = 1.f / (e0 + e1 + e2 + e3);
            const float* v0 = &QKV[2][s * 4 + 0][0];
            const float* v1 = &QKV[2][s * 4 + 1][0];
            const float* v2 = &QKV[2][s * 4 + 2][0];
            const float* v3 = &QKV[2][s * 4 + 3][0];
            const float o0 = (e0 * v0[c0] + e1 * v1[c0] + e2 * v2[c0] + e3 * v3[c0]) * inv;
            const float o1 = (e0 * v0[c0+1] + e1 * v1[c0+1] + e2 * v2[c0+1] + e3 * v3[c0+1]) * inv;
            const u32 dw = (u32)f2bf(o0) | ((u32)f2bf(o1) << 16);
            const int kcol  = h * 64 + c0;       // K-dim of proj GEMM
            const int ktile = kcol >> 5, kk = kcol & 31;
            const int hi = (kk >> 3) & 1;
            const int j0 = ((kk & 7) >> 1) + (kk >= 16 ? 4 : 0);
            const long mtg = (long)blockIdx.x * 4 + (row64 >> 4);
            AOp[((mtg * 32 + ktile) << 8) + ((row64 & 15) + hi * 16) * 8 + j0] = dw;
        }
        __syncthreads();   // QKV/sc reused next head
    }
}

// ---------------------------------------------------------------------------
// Kernel 2: proj GEMM (attn_out @ W2) on WMMA, then mean over T + bias.
// Block = 256 threads, output tile 64x64. Grid = (1024, 16).
// W2 panel (128KB) staged in LDS once (kills 4x redundant global B reads);
// A streams from the L2-resident (128MB) packed attention output.
// ---------------------------------------------------------------------------
__global__ __launch_bounds__(256) void kProj(
    const u32* __restrict__ AOp, const u32* __restrict__ W2p,
    const float* __restrict__ bias, float* __restrict__ out)
{
    __shared__ u32   W2s[128 * 256];             // 4 ntiles x 32 ktiles x 1KB
    __shared__ float Y[64][64];
    const int tid = threadIdx.x, wave = tid >> 5, lane = tid & 31;
    const int mblk = blockIdx.x;                 // 0..1023 (64-row groups)
    const int nblk = blockIdx.y;                 // 0..15   (64-col groups)

    // stage the block's W2 panel into LDS (8192 uint4, 32 per thread)
    for (int it = 0; it < 32; ++it) {
        const int idx = tid + it * 256;          // uint4 index 0..8191
        const int f = idx >> 6, q = idx & 63;    // frag = nt*32 + kt
        const int nt = f >> 5, kt = f & 31;
        ((uint4*)W2s)[idx] =
            ((const uint4*)(W2p + (((long)(nblk * 4 + nt) * 32 + kt) << 8)))[q];
    }
    __syncthreads();

    // wave -> mtile = wave>>1, ntiles {2*(wave&1), +1}: A shared by 2 WMMAs
    const int mtile  = wave >> 1;
    const int ntb    = (wave & 1) * 2;
    const long mtg   = (long)mblk * 4 + mtile;

    v8f acc[2] = {};
    for (int kt = 0; kt < 32; ++kt) {
        FragU a, b[2];
        const uint4* ap = (const uint4*)(AOp + ((mtg * 32 + kt) << 8)) + lane * 2;
        a.q[0] = ap[0];
        a.q[1] = ap[1];
        if (kt + 1 < 32)
            __builtin_prefetch((const void*)(ap + 64), 0, 1);
        #pragma unroll
        for (int j = 0; j < 2; ++j) {
            const u32* bp = &W2s[(((ntb + j) * 32 + kt) << 8) + lane * 8];
            b[j].q[0] = *(const uint4*)bp;
            b[j].q[1] = *(const uint4*)(bp + 4);
            acc[j] = __builtin_amdgcn_wmma_f32_16x16x32_bf16(
                false, a.v, false, b[j].v, (short)0, acc[j], false, false);
        }
    }
    #pragma unroll
    for (int j = 0; j < 2; ++j) {
        const int col   = (ntb + j) * 16 + (lane & 15);
        const int rbase = mtile * 16 + (lane >> 4) * 8;
        #pragma unroll
        for (int r = 0; r < 8; ++r) Y[rbase + r][col] = acc[j][r];
    }
    __syncthreads();

    // mean over T=4 rows + bias -> out[B, D]
    for (int it = 0; it < 4; ++it) {
        const int item = tid + it * 256;         // 16 samples x 64 cols
        const int s = item >> 6, c = item & 63;
        const float v = 0.25f * (Y[s * 4 + 0][c] + Y[s * 4 + 1][c] +
                                 Y[s * 4 + 2][c] + Y[s * 4 + 3][c]) +
                        bias[nblk * 64 + c];
        out[((long)mblk * 16 + s) * Dn + nblk * 64 + c] = v;
    }
}

// ---------------------------------------------------------------------------
extern "C" void kernel_launch(void* const* d_in, const int* in_sizes, int n_in,
                              void* d_out, int out_size, void* d_ws, size_t ws_size,
                              hipStream_t stream) {
    const float* x     = (const float*)d_in[0];
    const float* pos   = (const float*)d_in[1];
    const float* Wqkv  = (const float*)d_in[2];
    const float* Aq    = (const float*)d_in[3];
    const float* Bq    = (const float*)d_in[4];
    const float* Av    = (const float*)d_in[5];
    const float* Bv    = (const float*)d_in[6];
    const float* Wproj = (const float*)d_in[7];
    const float* bias  = (const float*)d_in[8];
    const float* Ap    = (const float*)d_in[9];
    const float* Bp    = (const float*)d_in[10];
    float* out = (float*)d_out;

    unsigned char* ws = (unsigned char*)d_ws;
    u32* W1p = (u32*)(ws);                         //   6 MB packed bf16 W1
    u32* W2p = (u32*)(ws + (6ull  << 20));         //   2 MB packed bf16 W2
    u32* AOp = (u32*)(ws + (8ull  << 20));         // 128 MB packed bf16 attn-out

    kPack<<<1024, 256, 0, stream>>>(Wqkv, Aq, Bq, Av, Bv, Wproj, Ap, Bp, W1p, W2p);
    kAttn<<<1024, 256, 0, stream>>>(x, pos, W1p, AOp);
    kProj<<<dim3(1024, 16), 256, 0, stream>>>(AOp, W2p, bias, out);
}
